// Seq2Seq_73985106641148
// MI455X (gfx1250) — matrix operand order repaired and checked by reading, not compiled, not run on hardware
//
#include <hip/hip_runtime.h>
#include <hip/hip_bf16.h>

typedef __bf16 bf16;
typedef __attribute__((ext_vector_type(16))) __bf16 v16bf;
typedef __attribute__((ext_vector_type(8)))  __bf16 v8bf;
typedef __attribute__((ext_vector_type(8)))  float  v8f;

#define V_  32000
#define E_  512
#define B_  16
#define TS_ 64
#define TT_ 48
#define NEG_ (-1000000000.0f)

// ---------------------------------------------------------------- helpers

__device__ __forceinline__ float sigmoidf_(float x) {
    return 1.0f / (1.0f + __expf(-x));
}

__device__ __forceinline__ v8f vzero8f() {
    v8f z = {0.f, 0.f, 0.f, 0.f, 0.f, 0.f, 0.f, 0.f};
    return z;
}

// A fragment (16x32 bf16, MxK) from a row-major bf16 matrix.
// lanes 0-15: row=lane,    K = k0+0..15
// lanes16-31: row=lane-16, K = k0+16..31
__device__ __forceinline__ v16bf load_a_frag(const bf16* __restrict__ A, int lda,
                                             int m0, int k0, int lane) {
    int row = m0 + (lane & 15);
    int kb  = k0 + ((lane >> 4) << 4);
    const bf16* p = A + (long)row * lda + kb;
    union { v16bf v; v8bf h[2]; } u;
    u.h[0] = *(const v8bf*)(p);
    u.h[1] = *(const v8bf*)(p + 8);
    return u.v;
}

// B fragment from pre-packed layout: 32 lanes x 16 bf16 contiguous (512 elems/tile)
__device__ __forceinline__ v16bf load_b_frag(const bf16* __restrict__ P, int lane) {
    const bf16* p = P + lane * 16;
    union { v16bf v; v8bf h[2]; } u;
    u.h[0] = *(const v8bf*)(p);
    u.h[1] = *(const v8bf*)(p + 8);
    return u.v;
}

__device__ __forceinline__ v8f wmma_bf(v16bf a, v16bf b, v8f c) {
    return __builtin_amdgcn_wmma_f32_16x16x32_bf16(false, a, false, b,
                                                   (short)0, c, false, false);
}

// ---------------------------------------------------------------- prep kernels

// Pack W (row-major [O][I] fp32, optionally a concat of two sources split at
// `split` along I) into bf16 B-fragment-major: [(I/32)][(O/16)][32 lanes][16].
// k = kt*32 + (lane>=16 ? 16 : 0) + e ; n = nt*16 + (lane&15)
__global__ __launch_bounds__(256) void pack_b(const float* __restrict__ S0, int ld0,
                                              const float* __restrict__ S1, int ld1,
                                              int split, int O, int I,
                                              bf16* __restrict__ out) {
    long idx = (long)blockIdx.x * blockDim.x + threadIdx.x;
    long total = (long)O * I;
    if (idx >= total) return;
    int  e    = (int)(idx & 15);
    long t    = idx >> 4;
    int  lane = (int)(t & 31);
    t >>= 5;
    int  NT = O >> 4;
    int  nt = (int)(t % NT);
    long kt = t / NT;
    int  k  = (int)(kt * 32) + ((lane >> 4) << 4) + e;
    int  n  = nt * 16 + (lane & 15);
    float val = (k < split) ? S0[(long)n * ld0 + k]
                            : S1[(long)n * ld1 + (k - split)];
    out[idx] = (bf16)val;
}

__global__ __launch_bounds__(256) void gather_emb(const int* __restrict__ tok,
                                                  const float* __restrict__ emb,
                                                  bf16* __restrict__ out, int nTok) {
    long i = (long)blockIdx.x * blockDim.x + threadIdx.x;
    long total = (long)nTok * E_;
    if (i >= total) return;
    int r = (int)(i >> 9);
    int e = (int)(i & 511);
    out[i] = (bf16)emb[(long)tok[r] * E_ + e];
}

// ---------------------------------------------------------------- encoder

// Xg[b*64+t][2048] = x_bf16 @ packed(W_ih_enc): M=1024, N=2048, K=512
__global__ __launch_bounds__(256) void gemm_xg(const bf16* __restrict__ xb,
                                               const bf16* __restrict__ WihP,
                                               float* __restrict__ Xg) {
    int lane = threadIdx.x & 31;
    int id   = blockIdx.x * 8 + (threadIdx.x >> 5);
    int mt   = id >> 7, nt = id & 127;
    v8f acc = vzero8f();
    for (int kt = 0; kt < 16; ++kt) {
        v16bf a = load_a_frag(xb, 512, mt * 16, kt * 32, lane);
        v16bf b = load_b_frag(WihP + (long)(kt * 128 + nt) * 512, lane);
        acc = wmma_bf(a, b, acc);
    }
    int n  = nt * 16 + (lane & 15);
    int mb = (lane >> 4) << 3;
#pragma unroll
    for (int r = 0; r < 8; ++r)
        Xg[(long)(mt * 16 + mb + r) * 2048 + n] = acc[r];
}

// Persistent single-WG masked LSTM scan. 32 waves; wave w owns columns
// [16w,16w+16) of every gate. h kept in LDS as bf16 (A operand), c in regs.
__global__ __launch_bounds__(1024) void encoder_kernel(
    const float* __restrict__ Xg, const bf16* __restrict__ WhhP,
    const int* __restrict__ lens, float* __restrict__ hs,
    float* __restrict__ s_g, float* __restrict__ m_g) {
    __shared__ bf16 hsh[16 * 512];
    int tid = threadIdx.x, lane = tid & 31, wave = tid >> 5;
    for (int i = tid; i < 16 * 512; i += 1024) hsh[i] = (bf16)0.0f;
    int n  = wave * 16 + (lane & 15);
    int mb = (lane >> 4) << 3;
    int lr[8];
#pragma unroll
    for (int r = 0; r < 8; ++r) lr[r] = lens[mb + r];
    v8f creg = vzero8f();
    __syncthreads();

    for (int t = 0; t < TS_; ++t) {
        v8f acc[4];
#pragma unroll
        for (int gi = 0; gi < 4; ++gi) {
#pragma unroll
            for (int r = 0; r < 8; ++r) {
                int b = mb + r;
                acc[gi][r] = Xg[(long)(b * 64 + t) * 2048 + gi * 512 + n];
            }
        }
        for (int kt = 0; kt < 16; ++kt) {
            v16bf a = load_a_frag(hsh, 512, 0, kt * 32, lane);
#pragma unroll
            for (int gi = 0; gi < 4; ++gi) {
                v16bf b = load_b_frag(WhhP + (long)(kt * 128 + gi * 32 + wave) * 512, lane);
                acc[gi] = wmma_bf(a, b, acc[gi]);
            }
        }
        __syncthreads();  // all waves done reading h
#pragma unroll
        for (int r = 0; r < 8; ++r) {
            int b = mb + r;
            float iv = sigmoidf_(acc[0][r]);
            float fv = sigmoidf_(acc[1][r]);
            float gv = tanhf(acc[2][r]);
            float ov = sigmoidf_(acc[3][r]);
            float cn = fv * creg[r] + iv * gv;
            float hn = ov * tanhf(cn);
            if (t < lr[r]) {  // packed-sequence freeze past length
                creg[r] = cn;
                hsh[b * 512 + n] = (bf16)hn;
                hs[(long)(b * 64 + t) * 512 + n] = hn;  // zeros elsewhere (memset)
            }
        }
        __syncthreads();
    }
#pragma unroll
    for (int r = 0; r < 8; ++r) {
        int b = mb + r;
        s_g[b * 512 + n] = (float)hsh[b * 512 + n];
        m_g[b * 512 + n] = creg[r];
    }
}

__global__ __launch_bounds__(256) void att1_kernel(const float* __restrict__ hs,
                                                   const float* __restrict__ Wa1,
                                                   float* __restrict__ att1) {
    int g = blockIdx.x * 8 + (threadIdx.x >> 5);  // 0..1023 (b*64+t)
    int lane = threadIdx.x & 31;
    const float* row = hs + (long)g * 512;
    float s = 0.f;
    for (int i = lane; i < 512; i += 32) s += row[i] * Wa1[i];
    for (int m = 16; m; m >>= 1) s += __shfl_xor(s, m);
    if (lane == 0) att1[g] = tanhf(s);
}

// ---------------------------------------------------------------- decoder

// One recurrent step: attention + W_s + LSTM gates. 1 WG, 32 waves.
__global__ __launch_bounds__(1024) void dec_step(
    int t, const bf16* __restrict__ tgtb, const float* __restrict__ hs,
    const float* __restrict__ att1, const int* __restrict__ lens,
    const float* __restrict__ Wa2, float* __restrict__ s_g,
    float* __restrict__ m_g, bf16* __restrict__ sc,
    const bf16* __restrict__ WsP, const bf16* __restrict__ WcatP) {
    __shared__ bf16 u[16 * 1024];   // [y_t | context]
    __shared__ bf16 v[16 * 1024];   // [d | s_prev]  (early: aliases alpha/swa2)
    float* alpha = (float*)v;               // 16*64 f32, dead before v is written
    float* swa2  = ((float*)v) + 16 * 64;   // 16 f32
    int tid = threadIdx.x, lane = tid & 31, wave = tid >> 5;

    // p0: tanh(s @ W_a2^T) per batch row
    if (wave < 16) {
        float s = 0.f;
        for (int i = lane; i < 512; i += 32) s += s_g[wave * 512 + i] * Wa2[i];
        for (int m = 16; m; m >>= 1) s += __shfl_xor(s, m);
        if (lane == 0) swa2[wave] = tanhf(s);
    }
    __syncthreads();

    // p1: masked softmax over TS=64 per batch row
    if (wave < 16) {
        int b = wave, L = lens[b];
        float ts = swa2[b];
        int  j1 = lane + 32;
        float e0 = (lane < L) ? att1[b * 64 + lane] * ts : NEG_;
        float e1 = (j1 < L) ? att1[b * 64 + j1] * ts : NEG_;
        float mx = fmaxf(e0, e1);
        for (int m = 16; m; m >>= 1) mx = fmaxf(mx, __shfl_xor(mx, m));
        float x0 = __expf(e0 - mx), x1 = __expf(e1 - mx);
        float sm = x0 + x1;
        for (int m = 16; m; m >>= 1) sm += __shfl_xor(sm, m);
        float inv = 1.0f / sm;
        alpha[b * 64 + lane] = x0 * inv;
        alpha[b * 64 + j1]   = x1 * inv;
    }
    __syncthreads();

    // p2: context c = alpha @ hs ; assemble u = [y_t | c]; stash c into sc hi half
    {
        int b = tid >> 6, eb = (tid & 63) * 8;
        float accv[8];
#pragma unroll
        for (int j = 0; j < 8; ++j) accv[j] = 0.f;
        const float* hrow = hs + (long)b * 64 * 512 + eb;
        for (int tt = 0; tt < 64; ++tt) {
            float a = alpha[b * 64 + tt];
#pragma unroll
            for (int j = 0; j < 8; ++j) accv[j] += a * hrow[(long)tt * 512 + j];
        }
#pragma unroll
        for (int j = 0; j < 8; ++j) {
            bf16 cv = (bf16)accv[j];
            u[b * 1024 + 512 + eb + j]  = cv;
            sc[b * 1024 + 512 + eb + j] = cv;
        }
        int i0 = tid * 8, bb = i0 >> 9, ee = i0 & 511;
#pragma unroll
        for (int j = 0; j < 8; ++j)
            u[bb * 1024 + ee + j] = tgtb[(long)(bb * 48 + t) * 512 + ee + j];
    }
    __syncthreads();

    // p3: d = u @ W_s^T  (N=512 -> 32 tiles, one per wave; K=1024)
    {
        v8f acc = vzero8f();
        for (int kt = 0; kt < 32; ++kt) {
            v16bf a = load_a_frag(u, 1024, 0, kt * 32, lane);
            v16bf b = load_b_frag(WsP + (long)(kt * 32 + wave) * 512, lane);
            acc = wmma_bf(a, b, acc);
        }
        int n = wave * 16 + (lane & 15), mb2 = (lane >> 4) << 3;
#pragma unroll
        for (int r = 0; r < 8; ++r) v[(mb2 + r) * 1024 + n] = (bf16)acc[r];
        // p3b: v hi half = s_prev
        int i0 = tid * 8, bb = i0 >> 9, ee = i0 & 511;
#pragma unroll
        for (int j = 0; j < 8; ++j)
            v[bb * 1024 + 512 + ee + j] = (bf16)s_g[bb * 512 + ee + j];
    }
    __syncthreads();

    // p4: gates = [d|s] @ [W_ih_dec|W_hh_dec]^T ; LSTM cell update
    {
        v8f acc[4];
#pragma unroll
        for (int gi = 0; gi < 4; ++gi) acc[gi] = vzero8f();
        for (int kt = 0; kt < 32; ++kt) {
            v16bf a = load_a_frag(v, 1024, 0, kt * 32, lane);
#pragma unroll
            for (int gi = 0; gi < 4; ++gi) {
                v16bf b = load_b_frag(WcatP + (long)(kt * 128 + gi * 32 + wave) * 512, lane);
                acc[gi] = wmma_bf(a, b, acc[gi]);
            }
        }
        int n = wave * 16 + (lane & 15), mb2 = (lane >> 4) << 3;
#pragma unroll
        for (int r = 0; r < 8; ++r) {
            int b = mb2 + r;
            float iv = sigmoidf_(acc[0][r]);
            float fv = sigmoidf_(acc[1][r]);
            float gv = tanhf(acc[2][r]);
            float ov = sigmoidf_(acc[3][r]);
            float mn = fv * m_g[b * 512 + n] + iv * gv;
            float sn = ov * tanhf(mn);
            m_g[b * 512 + n] = mn;
            s_g[b * 512 + n] = sn;
            sc[b * 1024 + n] = (bf16)sn;   // [s_ | c] for the vocab projection
        }
    }
}

// logits = sc @ W_y^T : M=16, N=32000, K=1024. 2000 N-tiles, 1 per wave.
__global__ __launch_bounds__(256) void logits_kernel(const bf16* __restrict__ sc,
                                                     const bf16* __restrict__ WyP,
                                                     float* __restrict__ logits) {
    int lane = threadIdx.x & 31;
    int nt   = blockIdx.x * 8 + (threadIdx.x >> 5);  // 0..1999
    const bf16* bp = WyP + (long)nt * 512;
    v8f acc = vzero8f();
    for (int kt = 0; kt < 32; ++kt) {
        if (kt + 1 < 32)
            __builtin_prefetch(bp + (long)(kt + 1) * 2000 * 512, 0, 1);
        v16bf a = load_a_frag(sc, 1024, 0, kt * 32, lane);
        v16bf b = load_b_frag(bp + (long)kt * 2000 * 512, lane);
        acc = wmma_bf(a, b, acc);
    }
    int n = nt * 16 + (lane & 15), mb = (lane >> 4) << 3;
#pragma unroll
    for (int r = 0; r < 8; ++r)
        logits[(long)(mb + r) * V_ + n] = acc[r];
}

__global__ __launch_bounds__(512) void softmax_kernel(int t,
                                                      const float* __restrict__ logits,
                                                      float* __restrict__ out) {
    int lane = threadIdx.x & 31, b = threadIdx.x >> 5;  // 16 waves, one row each
    const float* row = logits + (long)b * V_;
    float mx = -3.4e38f;
    for (int i = lane; i < V_; i += 32) mx = fmaxf(mx, row[i]);
    for (int m = 16; m; m >>= 1) mx = fmaxf(mx, __shfl_xor(mx, m));
    float sm = 0.f;
    for (int i = lane; i < V_; i += 32) sm += __expf(row[i] - mx);
    for (int m = 16; m; m >>= 1) sm += __shfl_xor(sm, m);
    float inv = 1.0f / sm;
    float* orow = out + ((long)b * TT_ + t) * V_;
    for (int i = lane; i < V_; i += 32) orow[i] = __expf(row[i] - mx) * inv;
}

// ---------------------------------------------------------------- launch

extern "C" void kernel_launch(void* const* d_in, const int* in_sizes, int n_in,
                              void* d_out, int out_size, void* d_ws, size_t ws_size,
                              hipStream_t stream) {
    const int*   mr    = (const int*)d_in[0];
    const int*   ref   = (const int*)d_in[1];
    const int*   lens  = (const int*)d_in[2];
    const float* emb   = (const float*)d_in[3];
    const float* Wih_e = (const float*)d_in[4];
    const float* Whh_e = (const float*)d_in[5];
    const float* Wih_d = (const float*)d_in[6];
    const float* Whh_d = (const float*)d_in[7];
    const float* Wy    = (const float*)d_in[8];
    const float* Ws    = (const float*)d_in[9];
    const float* Wa1   = (const float*)d_in[10];
    const float* Wa2   = (const float*)d_in[11];
    float* out = (float*)d_out;

    char* base = (char*)d_ws;
    size_t off = 0;
    auto alloc = [&](size_t bytes) -> char* {
        char* r = base + off;
        off = (off + bytes + 255) & ~(size_t)255;
        return r;
    };
    bf16*  WihEP = (bf16*)alloc((size_t)2048 * 512 * 2);
    bf16*  WhhEP = (bf16*)alloc((size_t)2048 * 512 * 2);
    bf16*  WcatP = (bf16*)alloc((size_t)2048 * 1024 * 2);
    bf16*  WsP   = (bf16*)alloc((size_t)512 * 1024 * 2);
    bf16*  WyP   = (bf16*)alloc((size_t)V_ * 1024 * 2);
    bf16*  x_bf  = (bf16*)alloc((size_t)B_ * TS_ * E_ * 2);
    bf16*  tgt_bf= (bf16*)alloc((size_t)B_ * TT_ * E_ * 2);
    float* Xg    = (float*)alloc((size_t)B_ * TS_ * 2048 * 4);
    float* hs    = (float*)alloc((size_t)B_ * TS_ * E_ * 4);
    float* att1  = (float*)alloc((size_t)B_ * TS_ * 4);
    float* s_g   = (float*)alloc((size_t)B_ * E_ * 4);
    float* m_g   = (float*)alloc((size_t)B_ * E_ * 4);
    bf16*  sc    = (bf16*)alloc((size_t)B_ * 1024 * 2);
    float* logit = (float*)alloc((size_t)B_ * V_ * 4);
    if (off > ws_size) return;  // scratch too small; bail deterministically

    // --- pack weights to bf16 B-fragment layout (once per call, parallel) ---
    {
        long n;
        n = 2048L * 512;
        pack_b<<<(int)((n + 255) / 256), 256, 0, stream>>>(Wih_e, 512, Wih_e, 512, 512, 2048, 512, WihEP);
        pack_b<<<(int)((n + 255) / 256), 256, 0, stream>>>(Whh_e, 512, Whh_e, 512, 512, 2048, 512, WhhEP);
        n = 2048L * 1024;
        pack_b<<<(int)((n + 255) / 256), 256, 0, stream>>>(Wih_d, 512, Whh_d, 512, 512, 2048, 1024, WcatP);
        n = 512L * 1024;
        pack_b<<<(int)((n + 255) / 256), 256, 0, stream>>>(Ws, 1024, Ws, 1024, 1024, 512, 1024, WsP);
        n = (long)V_ * 1024;
        pack_b<<<(int)((n + 255) / 256), 256, 0, stream>>>(Wy, 1024, Wy, 1024, 1024, V_, 1024, WyP);
    }
    // --- embedding gathers ---
    {
        long n = (long)B_ * TS_ * E_;
        gather_emb<<<(int)((n + 255) / 256), 256, 0, stream>>>(mr, emb, x_bf, B_ * TS_);
        n = (long)B_ * TT_ * E_;
        gather_emb<<<(int)((n + 255) / 256), 256, 0, stream>>>(ref, emb, tgt_bf, B_ * TT_);
    }
    hipMemsetAsync(hs, 0, (size_t)B_ * TS_ * E_ * 4, stream);

    // --- encoder ---
    gemm_xg<<<1024, 256, 0, stream>>>(x_bf, WihEP, Xg);
    encoder_kernel<<<1, 1024, 0, stream>>>(Xg, WhhEP, lens, hs, s_g, m_g);
    att1_kernel<<<128, 256, 0, stream>>>(hs, Wa1, att1);

    // --- decoder: 48 sequential steps ---
    for (int t = 0; t < TT_; ++t) {
        dec_step<<<1, 1024, 0, stream>>>(t, tgt_bf, hs, att1, lens, Wa2,
                                         s_g, m_g, sc, WsP, WcatP);
        logits_kernel<<<250, 256, 0, stream>>>(sc, WyP, logit);
        softmax_kernel<<<1, 512, 0, stream>>>(t, logit, out);
    }
    (void)in_sizes; (void)n_in; (void)out_size;
}
